// GCN_14688788152987
// MI455X (gfx1250) — matrix operand order repaired and checked by reading, not compiled, not run on hardware
//
#include <hip/hip_runtime.h>
#include <hip/hip_bf16.h>
#include <stdint.h>
#include <stddef.h>

#define NN 100000
#define NE 1600000
#define LN_EPS 1e-5f

typedef __attribute__((ext_vector_type(16))) __bf16 bf16x16;
typedef __attribute__((ext_vector_type(8)))  __bf16 bf16x8;
typedef __attribute__((ext_vector_type(8)))  float  f32x8;

union ABfrag { bf16x16 v; bf16x8 h8[2]; };

// ---------------- utility kernels ----------------

__global__ void zero_f32_kernel(float* __restrict__ p, int n4) {
  int i = blockIdx.x * blockDim.x + threadIdx.x;
  if (i < n4) ((float4*)p)[i] = make_float4(0.f, 0.f, 0.f, 0.f);
}

__global__ void degree_kernel(const int* __restrict__ src, const int* __restrict__ dst,
                              float* __restrict__ outdeg, float* __restrict__ indeg, int nE) {
  int e = blockIdx.x * blockDim.x + threadIdx.x;
  if (e < nE) {
    atomicAdd(&outdeg[src[e]], 1.0f);
    atomicAdd(&indeg[dst[e]], 1.0f);
  }
}

__global__ void norm_kernel(const float* __restrict__ outdeg, const float* __restrict__ indeg,
                            float* __restrict__ onorm, float* __restrict__ inorm, int n) {
  int i = blockIdx.x * blockDim.x + threadIdx.x;
  if (i < n) {
    onorm[i] = rsqrtf(fmaxf(outdeg[i], 1.0f));
    inorm[i] = rsqrtf(fmaxf(indeg[i], 1.0f));
  }
}

// W (K x N, row-major f32)  ->  Wt (Npad x K, row-major bf16), zero pad n >= N
__global__ void wt_bf16_kernel(const float* __restrict__ W, __bf16* __restrict__ Wt,
                               int K, int N, int Npad) {
  int idx = blockIdx.x * blockDim.x + threadIdx.x;
  if (idx < Npad * K) {
    int n = idx / K;
    int k = idx - n * K;
    float v = (n < N) ? W[(size_t)k * N + n] : 0.0f;
    Wt[idx] = (__bf16)v;
  }
}

// Xb[i,k] = bf16(X[i,k] * onorm[i]),  D = row width
__global__ void scale_bf16_kernel(const float* __restrict__ X, const float* __restrict__ onorm,
                                  __bf16* __restrict__ Xb, int total, int D) {
  int idx = blockIdx.x * blockDim.x + threadIdx.x;
  if (idx < total) {
    int i = idx / D;
    Xb[idx] = (__bf16)(X[idx] * onorm[i]);
  }
}

// ---------------- WMMA GEMM:  Y(Mn x N, f32) = X(Mn x K, bf16) @ Wt^T  ----------------
// Wt is stored N-major (Npad x K) so each lane's B fragment is contiguous in K.
// One wave32 computes a 16 x (16*NT) strip: the A fragment is loaded once per
// k-step and reused for NT v_wmma_f32_16x16x32_bf16 issues (A traffic / NT,
// load:wmma ratio 2+2*NT : NT).
template <int NT>
__global__ void gemm_wmma_kernel(const __bf16* __restrict__ X, const __bf16* __restrict__ Wt,
                                 float* __restrict__ Y, int Mn, int K, int N) {
  int wave = (blockIdx.x * blockDim.x + threadIdx.x) >> 5;
  int lane = threadIdx.x & 31;
  int mtiles  = Mn >> 4;
  int ngroups = (N >> 4) / NT;
  if (wave >= mtiles * ngroups) return;
  int tm = wave / ngroups;
  int tg = wave - tm * ngroups;

  int  l16 = lane & 15;
  bool hi  = lane >= 16;
  // A fragment: row m = lane&15; lanes 0-15 carry K [kb,kb+8)+[kb+16,kb+24),
  // lanes 16-31 the same shifted by 8 (ISA 7.12.2, 16-bit A 16x32).
  const __bf16* Arow = X + (size_t)(tm * 16 + l16) * K;
  int akoff = hi ? 8 : 0;
  // B fragment: column n = lane&15 (row n of Wt); lanes 0-15 K [kb,kb+8)+[kb+16,kb+24),
  // lanes 16-31 the same shifted by 8.
  int bkoff = hi ? 8 : 0;
  const __bf16* Brow[NT];
#pragma unroll
  for (int t = 0; t < NT; ++t)
    Brow[t] = Wt + (size_t)((tg * NT + t) * 16 + l16) * K;

  f32x8 acc[NT] = {};

  for (int kb = 0; kb < K; kb += 32) {
    ABfrag a;
    a.h8[0] = *(const bf16x8*)(Arow + kb + akoff);
    a.h8[1] = *(const bf16x8*)(Arow + kb + akoff + 16);
#pragma unroll
    for (int t = 0; t < NT; ++t) {
      ABfrag b;
      b.h8[0] = *(const bf16x8*)(Brow[t] + kb + bkoff);
      b.h8[1] = *(const bf16x8*)(Brow[t] + kb + bkoff + 16);
      acc[t] = __builtin_amdgcn_wmma_f32_16x16x32_bf16(false, a.v, false, b.v,
                                                       (short)0, acc[t], false, false);
    }
  }

  // C layout: VGPR r holds row m = r + 8*(lane>=16), col n = lane&15.
  int mbase = tm * 16 + (hi ? 8 : 0);
#pragma unroll
  for (int t = 0; t < NT; ++t) {
    float* Yp = Y + (size_t)mbase * N + (tg * NT + t) * 16 + l16;
#pragma unroll
    for (int r = 0; r < 8; ++r) Yp[(size_t)r * N] = acc[t][r];
  }
}

// ---------------- SpMM: M[dst] += Y[src]  (atomic f32 scatter-add) ----------------
__global__ void spmm_kernel(const float* __restrict__ Y, const int* __restrict__ src,
                            const int* __restrict__ dst, float* __restrict__ Mbuf,
                            int nE, int D) {
  int chunks = D >> 2;
  int idx = blockIdx.x * blockDim.x + threadIdx.x;
  if (idx >= nE * chunks) return;
  int e = idx / chunks;
  int c = idx - e * chunks;
  int s = src[e];
  int d = dst[e];
  float4 v = *(const float4*)(Y + (size_t)s * D + c * 4);
  float* mp = Mbuf + (size_t)d * D + c * 4;
  atomicAdd(mp + 0, v.x);
  atomicAdd(mp + 1, v.y);
  atomicAdd(mp + 2, v.z);
  atomicAdd(mp + 3, v.w);
}

// ---------------- fused in_norm+bias, LayerNorm, ReLU, out_norm, bf16 cast ----------------
// One wave32 per node, D = 128, 4 values per lane, wave32 shuffle reductions.
__global__ void ln_relu_kernel(const float* __restrict__ Mbuf, const float* __restrict__ inorm,
                               const float* __restrict__ bias, const float* __restrict__ gamma,
                               const float* __restrict__ beta, const float* __restrict__ onorm,
                               __bf16* __restrict__ Hb, int nN) {
  int node = (blockIdx.x * blockDim.x + threadIdx.x) >> 5;
  int lane = threadIdx.x & 31;
  if (node >= nN) return;
  int j = lane * 4;
  float inn = inorm[node];
  const float* mp = Mbuf + (size_t)node * 128 + j;
  float x0 = mp[0] * inn + bias[j + 0];
  float x1 = mp[1] * inn + bias[j + 1];
  float x2 = mp[2] * inn + bias[j + 2];
  float x3 = mp[3] * inn + bias[j + 3];

  float s = x0 + x1 + x2 + x3;
  for (int o = 16; o; o >>= 1) s += __shfl_xor(s, o, 32);
  float mu = s * (1.0f / 128.0f);

  float d0 = x0 - mu, d1 = x1 - mu, d2 = x2 - mu, d3 = x3 - mu;
  float vs = d0 * d0 + d1 * d1 + d2 * d2 + d3 * d3;
  for (int o = 16; o; o >>= 1) vs += __shfl_xor(vs, o, 32);
  float rstd = rsqrtf(vs * (1.0f / 128.0f) + LN_EPS);

  float on = onorm[node];
  float y0 = fmaxf(gamma[j + 0] * d0 * rstd + beta[j + 0], 0.0f) * on;
  float y1 = fmaxf(gamma[j + 1] * d1 * rstd + beta[j + 1], 0.0f) * on;
  float y2 = fmaxf(gamma[j + 2] * d2 * rstd + beta[j + 2], 0.0f) * on;
  float y3 = fmaxf(gamma[j + 3] * d3 * rstd + beta[j + 3], 0.0f) * on;

  __bf16* hp = Hb + (size_t)node * 128 + j;
  hp[0] = (__bf16)y0; hp[1] = (__bf16)y1; hp[2] = (__bf16)y2; hp[3] = (__bf16)y3;
}

// ---------------- final: out[i, j<40] = M2[i, j (of 48)] * inorm[i] + b2[j] ----------------
__global__ void final_kernel(const float* __restrict__ M2, const float* __restrict__ inorm,
                             const float* __restrict__ b2, float* __restrict__ out, int nN) {
  int idx = blockIdx.x * blockDim.x + threadIdx.x;
  if (idx >= nN * 40) return;
  int i = idx / 40;
  int j = idx - i * 40;
  out[idx] = M2[(size_t)i * 48 + j] * inorm[i] + b2[j];
}

// ---------------- host orchestration ----------------

static inline int cdiv(long a, int b) { return (int)((a + b - 1) / b); }

extern "C" void kernel_launch(void* const* d_in, const int* in_sizes, int n_in,
                              void* d_out, int out_size, void* d_ws, size_t ws_size,
                              hipStream_t stream) {
  const float* feat = (const float*)d_in[0];
  const int*   src  = (const int*)d_in[1];
  const int*   dst  = (const int*)d_in[2];
  const float* W0   = (const float*)d_in[3];
  const float* b0   = (const float*)d_in[4];
  const float* g0   = (const float*)d_in[5];
  const float* be0  = (const float*)d_in[6];
  const float* W1   = (const float*)d_in[7];
  const float* b1   = (const float*)d_in[8];
  const float* g1   = (const float*)d_in[9];
  const float* be1  = (const float*)d_in[10];
  const float* W2   = (const float*)d_in[11];
  const float* b2   = (const float*)d_in[12];
  float* out = (float*)d_out;

  char* p = (char*)d_ws;
  auto alloc = [&](size_t bytes) -> char* {
    char* r = p;
    p += (bytes + 511) & ~(size_t)511;
    return r;
  };
  float*  outdeg = (float*)alloc((size_t)NN * 4);
  float*  indeg  = (float*)alloc((size_t)NN * 4);
  float*  onorm  = (float*)alloc((size_t)NN * 4);
  float*  inorm  = (float*)alloc((size_t)NN * 4);
  __bf16* W0t    = (__bf16*)alloc((size_t)128 * 256 * 2);
  __bf16* W1t    = (__bf16*)alloc((size_t)128 * 128 * 2);
  __bf16* W2t    = (__bf16*)alloc((size_t)48 * 128 * 2);
  __bf16* Hb     = (__bf16*)alloc((size_t)NN * 256 * 2);  // Xb layer0, reused for H1b/H2b
  float*  Ybuf   = (float*)alloc((size_t)NN * 128 * 4);   // GEMM output, reused per layer
  float*  Mbuf   = (float*)alloc((size_t)NN * 128 * 4);   // aggregation buffer, reused

  const int B = 256;

  // degrees + norms
  zero_f32_kernel<<<cdiv(NN / 4, B), B, 0, stream>>>(outdeg, NN / 4);
  zero_f32_kernel<<<cdiv(NN / 4, B), B, 0, stream>>>(indeg, NN / 4);
  degree_kernel<<<cdiv(NE, B), B, 0, stream>>>(src, dst, outdeg, indeg, NE);
  norm_kernel<<<cdiv(NN, B), B, 0, stream>>>(outdeg, indeg, onorm, inorm, NN);

  // weights -> bf16 transposed (N-major), W2 padded 40 -> 48
  wt_bf16_kernel<<<cdiv(128 * 256, B), B, 0, stream>>>(W0, W0t, 256, 128, 128);
  wt_bf16_kernel<<<cdiv(128 * 128, B), B, 0, stream>>>(W1, W1t, 128, 128, 128);
  wt_bf16_kernel<<<cdiv(48 * 128, B), B, 0, stream>>>(W2, W2t, 128, 40, 48);

  // ---- layer 0 ----
  scale_bf16_kernel<<<cdiv((long)NN * 256, B), B, 0, stream>>>(feat, onorm, Hb, NN * 256, 256);
  {
    int waves = (NN / 16) * ((128 / 16) / 4);  // 16x64 strip per wave
    gemm_wmma_kernel<4><<<cdiv((long)waves * 32, B), B, 0, stream>>>(Hb, W0t, Ybuf, NN, 256, 128);
  }
  zero_f32_kernel<<<cdiv(NN * 128 / 4, B), B, 0, stream>>>(Mbuf, NN * 128 / 4);
  spmm_kernel<<<cdiv((long)NE * 32, B), B, 0, stream>>>(Ybuf, src, dst, Mbuf, NE, 128);
  ln_relu_kernel<<<cdiv((long)NN * 32, B), B, 0, stream>>>(Mbuf, inorm, b0, g0, be0, onorm, Hb, NN);

  // ---- layer 1 ----
  {
    int waves = (NN / 16) * ((128 / 16) / 4);
    gemm_wmma_kernel<4><<<cdiv((long)waves * 32, B), B, 0, stream>>>(Hb, W1t, Ybuf, NN, 128, 128);
  }
  zero_f32_kernel<<<cdiv(NN * 128 / 4, B), B, 0, stream>>>(Mbuf, NN * 128 / 4);
  spmm_kernel<<<cdiv((long)NE * 32, B), B, 0, stream>>>(Ybuf, src, dst, Mbuf, NE, 128);
  ln_relu_kernel<<<cdiv((long)NN * 32, B), B, 0, stream>>>(Mbuf, inorm, b1, g1, be1, onorm, Hb, NN);

  // ---- layer 2 (N = 40 padded to 48, one 16x48 strip per wave) ----
  {
    int waves = (NN / 16) * ((48 / 16) / 3);
    gemm_wmma_kernel<3><<<cdiv((long)waves * 32, B), B, 0, stream>>>(Hb, W2t, Ybuf, NN, 128, 48);
  }
  zero_f32_kernel<<<cdiv(NN * 48 / 4, B), B, 0, stream>>>(Mbuf, NN * 48 / 4);
  spmm_kernel<<<cdiv((long)NE * 12, B), B, 0, stream>>>(Ybuf, src, dst, Mbuf, NE, 48);
  final_kernel<<<cdiv((long)NN * 40, B), B, 0, stream>>>(Mbuf, inorm, b2, out, NN);

  (void)in_sizes; (void)n_in; (void)out_size; (void)ws_size;
}
